// MultiHeadAttention_72688026518067
// MI455X (gfx1250) — matrix operand order repaired and checked by reading, not compiled, not run on hardware
//
#include <hip/hip_runtime.h>

// ---------------------------------------------------------------------------
// MultiHeadAttention for MI455X (gfx1250):
//  - bf16 WMMA (v_wmma_f32_16x16x32_bf16), f32 accumulate
//  - GEMMs: Tensor Data Mover (tensor_load_to_lds) double-buffered staging,
//    TENSORcnt-tracked, overlapping DMA of tile k+1 with WMMA of tile k
//  - flash-attention inner loop (scores never hit HBM)
// Shapes fixed by the reference: B=4, S=2048, E=768, H=12, D=64.
// ---------------------------------------------------------------------------

#define BB   4
#define SS   2048
#define EE   768
#define HH   12
#define DD   64
#define MTOT (BB * SS)      // 8192
#define ATT_SCALE 0.125f    // 1/sqrt(64)

typedef __attribute__((ext_vector_type(16))) __bf16 v16bf;
typedef __attribute__((ext_vector_type(8)))  float  v8f;
typedef unsigned int u32x4 __attribute__((ext_vector_type(4)));
typedef int          i32x8 __attribute__((ext_vector_type(8)));
typedef int          i32x4 __attribute__((ext_vector_type(4)));

union FragBF { v16bf v; uint4 q[2]; };

__device__ __forceinline__ unsigned short f2bf(float f) {
    unsigned int u = __float_as_uint(f);
    u += 0x7FFFu + ((u >> 16) & 1u);            // round-to-nearest-even
    return (unsigned short)(u >> 16);
}

__device__ __forceinline__ v8f wmma_bf16(const v16bf& a, const v16bf& b, const v8f& c) {
    return __builtin_amdgcn_wmma_f32_16x16x32_bf16(false, a, false, b,
                                                   (short)0, c, false, false);
}

// ---------------------------------------------------------------------------
// TDM: DMA one 128x32-element bf16 tile (row stride = strideElems in memory)
// into LDS with 16B padding after every 64B row -> LDS row stride 80B (40 el).
// D# layout per CDNA5 ISA.
// Toolchain uses the 6-arg builtin: (g0, g1, g2, g3, g4, cpol).
// ---------------------------------------------------------------------------
__device__ __forceinline__ void tdm_load_tile(const unsigned short* gptr,
                                              unsigned short* lptr,
                                              int strideElems) {
    const unsigned lds = (unsigned)(size_t)lptr;      // low 32 bits = LDS offset
    const unsigned long long ga = (unsigned long long)(size_t)gptr;

    u32x4 g0;
    g0[0] = 1u;                                        // count=1, user descriptor
    g0[1] = lds;                                       // lds_addr
    g0[2] = (unsigned)ga;                              // global_addr[31:0]
    g0[3] = (unsigned)((ga >> 32) & 0x1FFFFFFu)        // global_addr[56:32]
          | (2u << 30);                                // type = 2 ("image")

    i32x8 g1;
    // data_size=2B (1), pad_enable, pad_interval=3 (16 DW), pad_amount=3 (4 DW)
    g1[0] = (1 << 16) | (1 << 20) | (3 << 22) | (3 << 25);
    const unsigned td0 = 0x7FFFFFFFu;                  // huge dims: no OOB clip
    const unsigned td1 = 0x7FFFFFFFu;
    g1[1] = (int)((td0 & 0xFFFFu) << 16);              // [63:48]=dim0 lo16
    g1[2] = (int)((td0 >> 16) | ((td1 & 0xFFFFu) << 16));
    g1[3] = (int)((td1 >> 16) | (32u << 16));          // tile_dim0 = 32
    g1[4] = 128;                                       // tile_dim1 = 128, tile_dim2 = 0
    g1[5] = strideElems;                               // tensor_dim0_stride lo32
    g1[6] = 0;                                         // stride hi16 | dim1_stride lo16
    g1[7] = 0;
    i32x4 z4 = { 0, 0, 0, 0 };
    i32x8 z8 = { 0, 0, 0, 0, 0, 0, 0, 0 };
    __builtin_amdgcn_tensor_load_to_lds(g0, g1, z4, z4, z8, 0);
}

// ---------------------------------------------------------------------------
// f32 -> bf16 elementwise (vectorized x4)
// ---------------------------------------------------------------------------
__global__ __launch_bounds__(256) void cvt_kernel(const float* __restrict__ in,
                                                  unsigned short* __restrict__ out,
                                                  int n4) {
    int i = blockIdx.x * 256 + threadIdx.x;
    if (i < n4) {
        float4 v = ((const float4*)in)[i];
        unsigned int lo = (unsigned)f2bf(v.x) | ((unsigned)f2bf(v.y) << 16);
        unsigned int hi = (unsigned)f2bf(v.z) | ((unsigned)f2bf(v.w) << 16);
        ((uint2*)out)[i] = make_uint2(lo, hi);
    }
}

// ---------------------------------------------------------------------------
// W[K,N] f32 -> Wt[N,K] bf16 (32x32 LDS tile transpose); K=N=768
// ---------------------------------------------------------------------------
__global__ __launch_bounds__(256) void wtrans_kernel(const float* __restrict__ W,
                                                     unsigned short* __restrict__ Wt) {
    __shared__ unsigned short tile[32][33];
    const int tx = threadIdx.x & 31;
    const int ty = threadIdx.x >> 5;           // 0..7
    const int n0 = blockIdx.x * 32;
    const int k0 = blockIdx.y * 32;
#pragma unroll
    for (int p = 0; p < 4; ++p)
        tile[ty + p * 8][tx] = f2bf(W[(size_t)(k0 + ty + p * 8) * EE + n0 + tx]);
    __syncthreads();
#pragma unroll
    for (int p = 0; p < 4; ++p)
        Wt[(size_t)(n0 + ty + p * 8) * EE + k0 + tx] = tile[tx][ty + p * 8];
}

// ---------------------------------------------------------------------------
// GEMM: Out = A[M,K](bf16) * Wt[N,K]^T(bf16) + bias[N]
// 128x128 block tile, 256 threads (8 waves), K-step 32.
// Both tiles DMA'd by the TDM into double-buffered padded LDS; wave 0 issues,
// waits TENSORcnt, workgroup barrier releases.  8 WMMAs / wave / K-step.
// OUT_MODE: 0 -> bf16 [B,H,S,D]; 1 -> bf16 [B,H,D,S]; 2 -> f32 row-major.
// ---------------------------------------------------------------------------
template <int OUT_MODE>
__global__ __launch_bounds__(256) void gemm_tdm_kernel(
    const unsigned short* __restrict__ A, const unsigned short* __restrict__ Wt,
    const float* __restrict__ bias, void* __restrict__ Out,
    int M, int N, int K)
{
    __shared__ __align__(16) unsigned short sA[2][128 * 40];
    __shared__ __align__(16) unsigned short sB[2][128 * 40];

    const int tid  = threadIdx.x;
    const int lane = tid & 31;
    const int wave = tid >> 5;
    const int wm   = wave >> 1;      // 0..3 : 32-row strip
    const int wn   = wave & 1;       // 0..1 : 64-col strip
    const int l16  = lane & 15;
    const int half = lane >> 4;
    const int bm   = blockIdx.x;
    const int bn   = blockIdx.y;
    const int nk   = K >> 5;

    v8f acc[2][4];
#pragma unroll
    for (int i = 0; i < 2; ++i)
#pragma unroll
        for (int j = 0; j < 4; ++j)
#pragma unroll
            for (int e = 0; e < 8; ++e) acc[i][j][e] = 0.0f;

    const unsigned short* Abase = A  + (size_t)(bm * 128) * K;
    const unsigned short* Bbase = Wt + (size_t)(bn * 128) * K;

    if (wave == 0) {
        tdm_load_tile(Abase, &sA[0][0], K);
        tdm_load_tile(Bbase, &sB[0][0], K);
        __builtin_amdgcn_s_wait_tensorcnt(0);
    }
    __syncthreads();

    for (int kt = 0; kt < nk; ++kt) {
        const int cur = kt & 1;
        const int nxt = cur ^ 1;
        if (wave == 0 && kt + 1 < nk) {       // DMA next tiles during compute
            tdm_load_tile(Abase + (kt + 1) * 32, &sA[nxt][0], K);
            tdm_load_tile(Bbase + (kt + 1) * 32, &sB[nxt][0], K);
        }

        FragBF afr[2], bfr[4];
#pragma unroll
        for (int i = 0; i < 2; ++i) {
            const unsigned short* pa = &sA[cur][(wm * 32 + i * 16 + l16) * 40 + half * 8];
            afr[i].q[0] = *(const uint4*)pa;        // K = half*8 + 0..7
            afr[i].q[1] = *(const uint4*)(pa + 16); // K = half*8 + 16..23
        }
#pragma unroll
        for (int j = 0; j < 4; ++j) {
            const unsigned short* pb = &sB[cur][(wn * 64 + j * 16 + l16) * 40 + half * 8];
            bfr[j].q[0] = *(const uint4*)pb;        // K = half*8 + 0..7
            bfr[j].q[1] = *(const uint4*)(pb + 16); // K = half*8 + 16..23
        }
#pragma unroll
        for (int i = 0; i < 2; ++i)
#pragma unroll
            for (int j = 0; j < 4; ++j)
                acc[i][j] = wmma_bf16(afr[i].v, bfr[j].v, acc[i][j]);

        if (wave == 0) __builtin_amdgcn_s_wait_tensorcnt(0);
        __syncthreads();
    }

    // ---- epilogue (OUT_MODE resolved at compile time) ----
#pragma unroll
    for (int j = 0; j < 4; ++j) {
        int n = bn * 128 + wn * 64 + j * 16 + l16;
        float bv = bias[n];
#pragma unroll
        for (int i = 0; i < 2; ++i) {
            int mbase = bm * 128 + wm * 32 + i * 16 + half * 8;
#pragma unroll
            for (int e = 0; e < 8; ++e) {
                float val = acc[i][j][e] + bv;
                int m = mbase + e;
                if (OUT_MODE == 2) {
                    ((float*)Out)[(size_t)m * N + n] = val;
                } else {
                    int b = m >> 11;      // m / S
                    int s = m & 2047;
                    int h = n >> 6;       // n / D
                    int d = n & 63;
                    unsigned short x = f2bf(val);
                    if (OUT_MODE == 0)    // [B,H,S,D]
                        ((unsigned short*)Out)[(((size_t)b * HH + h) * SS + s) * DD + d] = x;
                    else                  // [B,H,D,S]
                        ((unsigned short*)Out)[(((size_t)b * HH + h) * DD + d) * SS + s] = x;
                }
            }
        }
    }
}

// ---------------------------------------------------------------------------
// Flash attention: grid (S/64, B*H), 128 threads = 4 waves.
// Wave w owns query rows [qt*64 + w*16, +16) x full D=64.
// Q,Kt: bf16 [B,H,S,D]; Vt: bf16 [B,H,D,S]; ctx: bf16 [B,S,E].
// ---------------------------------------------------------------------------
__global__ __launch_bounds__(128) void attn_kernel(
    const unsigned short* __restrict__ Q,
    const unsigned short* __restrict__ Kt,
    const unsigned short* __restrict__ Vt,
    const int* __restrict__ mask,
    unsigned short* __restrict__ ctx)
{
    __shared__ __align__(16) unsigned short sP[4][16][72];  // per-wave P tile

    const int tid  = threadIdx.x;
    const int lane = tid & 31;
    const int wave = tid >> 5;
    const int l16  = lane & 15;
    const int half = lane >> 4;
    const int qt   = blockIdx.x;
    const int bh   = blockIdx.y;
    const int b    = bh / HH;
    const int h    = bh - b * HH;
    const size_t base = (size_t)bh * SS * DD;

    FragBF qf[2];
    {
        const unsigned short* qrow = Q + base + (size_t)(qt * 64 + wave * 16 + l16) * DD;
#pragma unroll
        for (int dstep = 0; dstep < 2; ++dstep) {
            const unsigned short* p = qrow + dstep * 32 + half * 8;
            qf[dstep].q[0] = *(const uint4*)p;
            qf[dstep].q[1] = *(const uint4*)(p + 16);
        }
    }

    v8f o[4];
#pragma unroll
    for (int t = 0; t < 4; ++t)
#pragma unroll
        for (int e = 0; e < 8; ++e) o[t][e] = 0.0f;
    float mrow[8], lrow[8];
#pragma unroll
    for (int e = 0; e < 8; ++e) { mrow[e] = -3.0e38f; lrow[e] = 0.0f; }

    for (int k0 = 0; k0 < SS; k0 += 64) {
        // ---- S = Q @ K^T ----
        v8f sc[4];
#pragma unroll
        for (int j = 0; j < 4; ++j) {
#pragma unroll
            for (int e = 0; e < 8; ++e) sc[j][e] = 0.0f;
            const unsigned short* kp = Kt + base + (size_t)(k0 + j * 16 + l16) * DD + half * 8;
#pragma unroll
            for (int dstep = 0; dstep < 2; ++dstep) {
                FragBF kf;
                const unsigned short* p = kp + dstep * 32;
                kf.q[0] = *(const uint4*)p;
                kf.q[1] = *(const uint4*)(p + 16);
                sc[j] = wmma_bf16(qf[dstep].v, kf.v, sc[j]);
            }
        }
        // ---- scale + padding-mask bias ----
        float bias_j[4];
#pragma unroll
        for (int j = 0; j < 4; ++j) {
            int mv = mask[(size_t)b * SS + k0 + j * 16 + l16];
            bias_j[j] = mv ? 0.0f : -1.0e9f;
        }
        // ---- online softmax (rows live in 16-lane halves of C layout) ----
#pragma unroll
        for (int e = 0; e < 8; ++e) {
            float mx = -3.0e38f;
#pragma unroll
            for (int j = 0; j < 4; ++j) {
                float s = sc[j][e] * ATT_SCALE + bias_j[j];
                sc[j][e] = s;
                mx = fmaxf(mx, s);
            }
            mx = fmaxf(mx, __shfl_xor(mx, 1));
            mx = fmaxf(mx, __shfl_xor(mx, 2));
            mx = fmaxf(mx, __shfl_xor(mx, 4));
            mx = fmaxf(mx, __shfl_xor(mx, 8));
            float mnew = fmaxf(mrow[e], mx);
            float fac  = __expf(mrow[e] - mnew);
            mrow[e] = mnew;
            float rs = 0.0f;
#pragma unroll
            for (int j = 0; j < 4; ++j) {
                float pe = __expf(sc[j][e] - mnew);
                sc[j][e] = pe;
                rs += pe;
            }
            rs += __shfl_xor(rs, 1);
            rs += __shfl_xor(rs, 2);
            rs += __shfl_xor(rs, 4);
            rs += __shfl_xor(rs, 8);
            lrow[e] = lrow[e] * fac + rs;
#pragma unroll
            for (int t = 0; t < 4; ++t) o[t][e] *= fac;
        }
        // ---- P (C layout) -> LDS -> A-fragment layout; same-wave DS in-order ----
#pragma unroll
        for (int j = 0; j < 4; ++j)
#pragma unroll
            for (int e = 0; e < 8; ++e)
                sP[wave][half * 8 + e][j * 16 + l16] = f2bf(sc[j][e]);
        // ---- O += P @ V ----
#pragma unroll
        for (int dstep = 0; dstep < 2; ++dstep) {
            FragBF pf;
            const unsigned short* pp = &sP[wave][l16][dstep * 32 + half * 8];
            pf.q[0] = *(const uint4*)pp;
            pf.q[1] = *(const uint4*)(pp + 16);
#pragma unroll
            for (int t = 0; t < 4; ++t) {
                FragBF vf;
                const unsigned short* vp = Vt + base + (size_t)(t * 16 + l16) * SS
                                         + k0 + dstep * 32 + half * 8;
                vf.q[0] = *(const uint4*)vp;
                vf.q[1] = *(const uint4*)(vp + 16);
                o[t] = wmma_bf16(pf.v, vf.v, o[t]);
            }
        }
    }

    // ---- normalize and store ctx (bf16 [B,S,E], E index = h*64 + d) ----
#pragma unroll
    for (int e = 0; e < 8; ++e) {
        float inv = lrow[e] > 0.0f ? 1.0f / lrow[e] : 0.0f;
        int s = qt * 64 + wave * 16 + half * 8 + e;
        unsigned short* crow = ctx + ((size_t)b * SS + s) * EE + h * DD;
#pragma unroll
        for (int t = 0; t < 4; ++t)
            crow[t * 16 + l16] = f2bf(o[t][e] * inv);
    }
}

// ---------------------------------------------------------------------------
extern "C" void kernel_launch(void* const* d_in, const int* in_sizes, int n_in,
                              void* d_out, int out_size, void* d_ws, size_t ws_size,
                              hipStream_t stream) {
    (void)in_sizes; (void)n_in; (void)out_size; (void)ws_size;
    const float* q    = (const float*)d_in[0];
    const float* k    = (const float*)d_in[1];
    const float* v    = (const float*)d_in[2];
    const int*   mask = (const int*)  d_in[3];
    const float* Wq   = (const float*)d_in[4];
    const float* bq   = (const float*)d_in[5];
    const float* Wk   = (const float*)d_in[6];
    const float* bk   = (const float*)d_in[7];
    const float* Wv   = (const float*)d_in[8];
    const float* bv   = (const float*)d_in[9];
    const float* Wo   = (const float*)d_in[10];
    const float* bo   = (const float*)d_in[11];

    const size_t NE = (size_t)MTOT * EE;     // 6,291,456
    const size_t WE = (size_t)EE * EE;       //   589,824
    unsigned short* qbf = (unsigned short*)d_ws;
    unsigned short* kbf = qbf + NE;
    unsigned short* vbf = kbf + NE;
    unsigned short* WqT = vbf + NE;
    unsigned short* WkT = WqT + WE;
    unsigned short* WvT = WkT + WE;
    unsigned short* WoT = WvT + WE;
    unsigned short* qh  = WoT + WE;          // [B,H,S,D]
    unsigned short* kh  = qh + NE;
    unsigned short* vt  = kh + NE;           // [B,H,D,S]
    unsigned short* cx  = vt + NE;           // [B,S,E]

    // ---- f32 -> bf16 inputs, transposed bf16 weights ----
    const int n4 = (int)(NE / 4);
    dim3 cgrid((n4 + 255) / 256);
    cvt_kernel<<<cgrid, 256, 0, stream>>>(q, qbf, n4);
    cvt_kernel<<<cgrid, 256, 0, stream>>>(k, kbf, n4);
    cvt_kernel<<<cgrid, 256, 0, stream>>>(v, vbf, n4);
    dim3 tgrid(EE / 32, EE / 32);
    wtrans_kernel<<<tgrid, 256, 0, stream>>>(Wq, WqT);
    wtrans_kernel<<<tgrid, 256, 0, stream>>>(Wk, WkT);
    wtrans_kernel<<<tgrid, 256, 0, stream>>>(Wv, WvT);
    wtrans_kernel<<<tgrid, 256, 0, stream>>>(Wo, WoT);

    // ---- QKV projections (TDM-staged WMMA GEMMs) ----
    dim3 ggrid(MTOT / 128, EE / 128);        // (64, 6)
    gemm_tdm_kernel<0><<<ggrid, 256, 0, stream>>>(qbf, WqT, bq, (void*)qh, MTOT, EE, EE);
    gemm_tdm_kernel<0><<<ggrid, 256, 0, stream>>>(kbf, WkT, bk, (void*)kh, MTOT, EE, EE);
    gemm_tdm_kernel<1><<<ggrid, 256, 0, stream>>>(vbf, WvT, bv, (void*)vt, MTOT, EE, EE);

    // ---- flash attention ----
    attn_kernel<<<dim3(SS / 64, BB * HH), 128, 0, stream>>>(qh, kh, vt, mask, cx);

    // ---- output projection -> d_out (f32) ----
    gemm_tdm_kernel<2><<<ggrid, 256, 0, stream>>>(cx, WoT, bo, d_out, MTOT, EE, EE);
}
